// MDCR_3246995275942
// MI455X (gfx1250) — matrix operand order repaired and checked by reading, not compiled, not run on hardware
//
#include <hip/hip_runtime.h>
#include <cstdint>

// ---------------------------------------------------------------------------
// MDCR fused pipeline for gfx1250 (MI455X), wave32.
//
// Kernel 1: fused {4x depthwise dilated 3x3 conv + bias + BN + ReLU} ->
//           {4x4 branch mix + BN + ReLU} -> split result into bf16 hi/lo and
//           store PRE-SWIZZLED into the WMMA B-fragment layout:
//             block(kc, ntile) of 32K x 16N = 1KB contiguous,
//             element offset = kc*N*32 + (n/16)*512 + lane*16 + (k%16)
//             where lane = (n%16) + 16*((k%32)/16).
// Kernel 2: GEMM out = ReLU(BN(Wo @ z + bo)) with v_wmma_f32_16x16x32_bf16,
//           bf16x2 split (A_hi*B_hi + A_hi*B_lo + A_lo*B_hi) for ~f32 accuracy.
//           B fragments load directly from global (pre-swizzled); A (Wo) is
//           staged per 32-K chunk into LDS in fragment order as bf16 hi/lo.
// ---------------------------------------------------------------------------

typedef __attribute__((ext_vector_type(16))) __bf16 v16bf;
typedef __attribute__((ext_vector_type(8)))  float  v8f;

#define N_PIX   73728      // 8*96*96 pixels
#define HW      9216       // 96*96
#define CQ      128
#define ZCHUNK  2359296    // N_PIX*32 elements per 32-wide K chunk

__device__ __forceinline__ unsigned short bf16_bits(float f) {
    __bf16 b = (__bf16)f;                       // RNE convert
    return __builtin_bit_cast(unsigned short, b);
}
__device__ __forceinline__ float bf16_val(unsigned short u) {
    __bf16 b = __builtin_bit_cast(__bf16, u);
    return (float)b;
}

// ---------------------------------------------------------------------------
// Kernel 1: depthwise convs + mix, write swizzled bf16 hi/lo z.
// grid = (N_PIX/256, 32), block = 256.  Thread (n, j): pixel n, K-range
// [16j, 16j+16) i.e. channel group c in [4j, 4j+4), 4 mix outputs each.
// ---------------------------------------------------------------------------
__global__ __launch_bounds__(256) void mdcr_dw_mix(
    const float* __restrict__ x,   const float* __restrict__ wdw,
    const float* __restrict__ bdw, const float* __restrict__ gdw,
    const float* __restrict__ bedw,const float* __restrict__ mdw,
    const float* __restrict__ vdw, const float* __restrict__ wsm,
    const float* __restrict__ bs,  const float* __restrict__ gs,
    const float* __restrict__ bes, const float* __restrict__ ms,
    const float* __restrict__ vs,
    unsigned short* __restrict__ zhi, unsigned short* __restrict__ zlo)
{
    const int n  = blockIdx.x * 256 + threadIdx.x;   // pixel index
    const int j  = blockIdx.y;                       // 0..31
    const int b  = n / HW;
    const int hw = n - b * HW;
    const int h  = hw / 96;
    const int w  = hw - h * 96;

    // mix-stage affine: z = relu((sum_i W4[o,i]*y_i + bs)*inv + (bes - ms*inv))
    float as_[4], bt_[4];
#pragma unroll
    for (int o = 0; o < 4; ++o) {
        float inv = gs[o] * rsqrtf(vs[o] + 1e-5f);
        as_[o] = inv;
        bt_[o] = bes[o] - ms[o] * inv + bs[o] * inv;
    }

    unsigned int hiw[8] = {0,0,0,0,0,0,0,0};
    unsigned int low[8] = {0,0,0,0,0,0,0,0};

#pragma unroll
    for (int cc = 0; cc < 4; ++cc) {
        const int c = j * 4 + cc;                    // 0..127
        float yr[4];
#pragma unroll
        for (int i = 0; i < 4; ++i) {
            const int r  = (i == 0) ? 1 : (i == 1) ? 6 : (i == 2) ? 12 : 18;
            const int ch = i * CQ + c;               // input channel 0..511
            const float* xp = x + (size_t)(b * 512 + ch) * HW;
            const float* wp = wdw + (size_t)ch * 9;
            float acc = 0.f;
#pragma unroll
            for (int ky = 0; ky < 3; ++ky) {
                const int hh = h + (ky - 1) * r;
                if (hh < 0 || hh >= 96) continue;
                const float* rowp = xp + hh * 96;
#pragma unroll
                for (int kx = 0; kx < 3; ++kx) {
                    const int ww = w + (kx - 1) * r;
                    if (ww < 0 || ww >= 96) continue;
                    acc = fmaf(wp[ky * 3 + kx], rowp[ww], acc);
                }
            }
            // (conv + bdw) -> BN -> ReLU, folded affine
            const float inv = gdw[ch] * rsqrtf(vdw[ch] + 1e-5f);
            const float bb  = bedw[ch] - mdw[ch] * inv + bdw[ch] * inv;
            yr[i] = fmaxf(fmaf(acc, inv, bb), 0.f);
        }
#pragma unroll
        for (int o = 0; o < 4; ++o) {
            float s = 0.f;
#pragma unroll
            for (int i = 0; i < 4; ++i) s = fmaf(wsm[o * 4 + i], yr[i], s);
            const float zv = fmaxf(fmaf(s, as_[o], bt_[o]), 0.f);
            const unsigned short hb = bf16_bits(zv);
            const unsigned short lb = bf16_bits(zv - bf16_val(hb));
            const int p = cc * 4 + o;                // 0..15 within K-group
            hiw[p >> 1] |= (unsigned int)hb << (16 * (p & 1));
            low[p >> 1] |= (unsigned int)lb << (16 * (p & 1));
        }
    }

    // swizzled store: one aligned 32B hi + 32B lo
    const size_t off = (size_t)(j >> 1) * ZCHUNK + (size_t)(n >> 4) * 512
                     + (size_t)(j & 1) * 256 + (size_t)(n & 15) * 16;
    uint4 h0 = {hiw[0], hiw[1], hiw[2], hiw[3]};
    uint4 h1 = {hiw[4], hiw[5], hiw[6], hiw[7]};
    uint4 l0 = {low[0], low[1], low[2], low[3]};
    uint4 l1 = {low[4], low[5], low[6], low[7]};
    *(uint4*)(zhi + off)     = h0;
    *(uint4*)(zhi + off + 8) = h1;
    *(uint4*)(zlo + off)     = l0;
    *(uint4*)(zlo + off + 8) = l1;
}

// ---------------------------------------------------------------------------
// Kernel 2: GEMM (M=512, K=512, N=73728) with v_wmma_f32_16x16x32_bf16.
// grid = (N_PIX/128, 512/128), block = 256 (8 waves, 2M x 4N wave grid).
// Wave tile 64M x 32N -> 8 accumulators; K chunked by 32; 24 WMMA per chunk.
// ---------------------------------------------------------------------------
__global__ __launch_bounds__(256) void mdcr_gemm(
    const unsigned short* __restrict__ zhi, const unsigned short* __restrict__ zlo,
    const float* __restrict__ wo, const float* __restrict__ bo,
    const float* __restrict__ go, const float* __restrict__ beo,
    const float* __restrict__ mo, const float* __restrict__ vo,
    float* __restrict__ out)
{
    // A tile in fragment order, row stride 48 shorts (96B, 32B-aligned frags)
    __shared__ unsigned short Ah[128 * 48];
    __shared__ unsigned short Al[128 * 48];
    __shared__ float alpha_s[128];
    __shared__ float beta_s[128];

    const int tid = threadIdx.x;
    const int m0  = blockIdx.y * 128;
    const int n0  = blockIdx.x * 128;

    if (tid < 128) {
        const int o = m0 + tid;
        const float inv = go[o] * rsqrtf(vo[o] + 1e-5f);
        alpha_s[tid] = inv;
        beta_s[tid]  = beo[o] - mo[o] * inv + bo[o] * inv;
    }

    const int wave  = tid >> 5;
    const int lane  = tid & 31;
    const int mg    = wave >> 2;      // 0..1 (64-row M group)
    const int ng    = wave & 3;       // 0..3 (32-col N group)
    const int l15   = lane & 15;
    const int lhalf = lane >> 4;      // 0/1

    v8f acc[4][2];
#pragma unroll
    for (int mt = 0; mt < 4; ++mt)
#pragma unroll
        for (int nt = 0; nt < 2; ++nt)
#pragma unroll
            for (int e = 0; e < 8; ++e) acc[mt][nt][e] = 0.f;

    for (int kc = 0; kc < 16; ++kc) {
        __syncthreads();
        // ---- stage A chunk (128 rows x 32 k) as bf16 hi/lo in fragment order
#pragma unroll
        for (int it = 0; it < 4; ++it) {
            const int q   = it * 256 + tid;      // quad id 0..1023
            const int row = q >> 3;
            const int kq  = (q & 7) << 2;        // k quad base 0..28
            const float4 wv = *(const float4*)(wo + (size_t)(m0 + row) * 512
                                               + (size_t)kc * 32 + kq);
            // k -> fragment position p: [K0..15 | K16..31]
            const int p0 = kq;
            const float vv[4] = {wv.x, wv.y, wv.z, wv.w};
            unsigned long long hq = 0ull, lq = 0ull;
#pragma unroll
            for (int e = 0; e < 4; ++e) {
                const unsigned short hb = bf16_bits(vv[e]);
                const unsigned short lb = bf16_bits(vv[e] - bf16_val(hb));
                hq |= (unsigned long long)hb << (16 * e);
                lq |= (unsigned long long)lb << (16 * e);
            }
            *(unsigned long long*)(&Ah[row * 48 + p0]) = hq;
            *(unsigned long long*)(&Al[row * 48 + p0]) = lq;
        }
        __syncthreads();

        // ---- B fragments straight from global (pre-swizzled): 32B/lane
        const size_t kcb = (size_t)kc * ZCHUNK;
        v16bf bhi[2], blo[2];
#pragma unroll
        for (int nt = 0; nt < 2; ++nt) {
            const int ncol  = n0 + ng * 32 + nt * 16;
            const size_t bo_ = kcb + (size_t)(ncol >> 4) * 512 + (size_t)lane * 16;
            bhi[nt] = *(const v16bf*)(zhi + bo_);
            blo[nt] = *(const v16bf*)(zlo + bo_);
        }
        if (kc < 15) {      // global_prefetch next chunk's B
            const size_t pf = kcb + ZCHUNK
                            + (size_t)((n0 + ng * 32) >> 4) * 512 + (size_t)lane * 16;
            __builtin_prefetch(zhi + pf, 0, 1);
            __builtin_prefetch(zlo + pf, 0, 1);
        }

        // ---- 24 WMMAs: bf16x2 split (drop lo*lo term)
#pragma unroll
        for (int mt = 0; mt < 4; ++mt) {
            const int row = mg * 64 + mt * 16 + l15;
            const v16bf ah = *(const v16bf*)(&Ah[row * 48 + lhalf * 16]);
            const v16bf al = *(const v16bf*)(&Al[row * 48 + lhalf * 16]);
#pragma unroll
            for (int nt = 0; nt < 2; ++nt) {
                acc[mt][nt] = __builtin_amdgcn_wmma_f32_16x16x32_bf16(
                    false, ah, false, bhi[nt], (short)0, acc[mt][nt], false, false);
                acc[mt][nt] = __builtin_amdgcn_wmma_f32_16x16x32_bf16(
                    false, ah, false, blo[nt], (short)0, acc[mt][nt], false, false);
                acc[mt][nt] = __builtin_amdgcn_wmma_f32_16x16x32_bf16(
                    false, al, false, bhi[nt], (short)0, acc[mt][nt], false, false);
            }
        }
    }
    __syncthreads();

    // ---- epilogue: BN affine + ReLU, store NCHW f32
#pragma unroll
    for (int nt = 0; nt < 2; ++nt) {
        const int n  = n0 + ng * 32 + nt * 16 + l15;
        const int b  = n / HW;
        const int hw = n - b * HW;
        float* obase = out + (size_t)b * 512 * HW + hw;
#pragma unroll
        for (int mt = 0; mt < 4; ++mt) {
#pragma unroll
            for (int r = 0; r < 8; ++r) {
                const int ml = mg * 64 + mt * 16 + r + lhalf * 8;  // C-frag row
                const float v = fmaxf(fmaf(acc[mt][nt][r], alpha_s[ml], beta_s[ml]), 0.f);
                obase[(size_t)(m0 + ml) * HW] = v;
            }
        }
    }
}

// ---------------------------------------------------------------------------
extern "C" void kernel_launch(void* const* d_in, const int* in_sizes, int n_in,
                              void* d_out, int out_size, void* d_ws, size_t ws_size,
                              hipStream_t stream)
{
    const float* x    = (const float*)d_in[0];
    const float* wdw  = (const float*)d_in[1];
    const float* bdw  = (const float*)d_in[2];
    const float* gdw  = (const float*)d_in[3];
    const float* bedw = (const float*)d_in[4];
    const float* mdw  = (const float*)d_in[5];
    const float* vdw  = (const float*)d_in[6];
    const float* wsm  = (const float*)d_in[7];
    const float* bs   = (const float*)d_in[8];
    const float* gs   = (const float*)d_in[9];
    const float* bes  = (const float*)d_in[10];
    const float* ms   = (const float*)d_in[11];
    const float* vs   = (const float*)d_in[12];
    const float* wo   = (const float*)d_in[13];
    const float* bo   = (const float*)d_in[14];
    const float* go   = (const float*)d_in[15];
    const float* beo  = (const float*)d_in[16];
    const float* mo   = (const float*)d_in[17];
    const float* vo   = (const float*)d_in[18];
    float* out = (float*)d_out;

    // workspace: z_hi and z_lo, each N_PIX*512 bf16 = 75.5 MB (151 MB total)
    unsigned short* zhi = (unsigned short*)d_ws;
    unsigned short* zlo = zhi + (size_t)N_PIX * 512;

    dim3 b1(256), g1(N_PIX / 256, 32);
    hipLaunchKernelGGL(mdcr_dw_mix, g1, b1, 0, stream,
                       x, wdw, bdw, gdw, bedw, mdw, vdw,
                       wsm, bs, gs, bes, ms, vs, zhi, zlo);

    dim3 b2(256), g2(N_PIX / 128, 512 / 128);
    hipLaunchKernelGGL(mdcr_gemm, g2, b2, 0, stream,
                       zhi, zlo, wo, bo, go, beo, mo, vo, out);
}
